// NestedPhysioNeuron_3934190043644
// MI455X (gfx1250) — matrix operand order repaired and checked by reading, not compiled, not run on hardware
//
#include <hip/hip_runtime.h>
#include <hip/hip_bf16.h>
#include <cstdint>

#define B_SZ    65536
#define D_INx   512
#define D_OUTx  512
#define HIDx    1024
#define BASE_LR 0.1f
#define LN_EPS  1e-5f

typedef __attribute__((ext_vector_type(16))) _Float16 v16h;
typedef __attribute__((ext_vector_type(8)))  _Float16 v8h;
typedef __attribute__((ext_vector_type(8)))  float    v8f;
typedef __attribute__((ext_vector_type(4))) unsigned int u32x4;
typedef __attribute__((ext_vector_type(8))) int          i32x8;
typedef __attribute__((ext_vector_type(4))) int          i32x4;

union HV { v16h v; v8h h[2]; };

__device__ __forceinline__ v8f wmma16x16x32(v16h a, v16h b, v8f c) {
  return __builtin_amdgcn_wmma_f32_16x16x32_f16(false, a, false, b, (short)0, c, false, false);
}

// A fragment (16x32 f16): lane m = L&15;
// elements 0-15 -> K = hi*16 + e
__device__ __forceinline__ v16h load_a_frag_lds(const _Float16* tile, int lane, int ldk) {
  const _Float16* p = tile + (lane & 15) * ldk + (lane >> 4) * 16;
  HV u;
  u.h[0] = *(const v8h*)(p);
  u.h[1] = *(const v8h*)(p + 8);
  return u.v;
}

// Same fragment loaded straight from a row-major f16 matrix in global memory.
__device__ __forceinline__ v16h load_a_frag_global(const _Float16* A, int m, int k0, int ldk, int lane) {
  const _Float16* p = A + (size_t)(m + (lane & 15)) * ldk + k0 + (lane >> 4) * 16;
  HV u;
  u.h[0] = *(const v8h*)(p);
  u.h[1] = *(const v8h*)(p + 8);
  return u.v;
}

// B fragment (32x16 f16) from row-major f16 W[n][k]: lane n = L&15 ; K = hi*16 + e
__device__ __forceinline__ v16h load_b_frag_global(const _Float16* W, int n, int k0, int ldk, int lane) {
  const _Float16* p = W + (size_t)(n + (lane & 15)) * ldk + k0 + (lane >> 4) * 16;
  HV u;
  u.h[0] = *(const v8h*)(p);
  u.h[1] = *(const v8h*)(p + 8);
  return u.v;
}

__device__ __forceinline__ v16h load_b_frag_lds(const _Float16* tile, int lane, int ldk) {
  const _Float16* p = tile + (lane & 15) * ldk + (lane >> 4) * 16;
  HV u;
  u.h[0] = *(const v8h*)(p);
  u.h[1] = *(const v8h*)(p + 8);
  return u.v;
}

__device__ __forceinline__ float sigmoidf(float x) { return 1.0f / (1.0f + __expf(-x)); }

// ---------------------------------------------------------------- TDM: DMA a 2D f16 tile (row-major) into LDS
__device__ __forceinline__ void tdm_load_tile_f16(unsigned lds_byte_addr, const _Float16* gsrc,
                                                  unsigned dim0, unsigned dim1, unsigned stride0,
                                                  unsigned tile_w, unsigned tile_h) {
  const unsigned long long ga = (unsigned long long)(uintptr_t)gsrc;
  u32x4 g0;
  g0[0] = 1u;                                                // count=1 (valid user descriptor)
  g0[1] = lds_byte_addr;                                     // lds_addr
  g0[2] = (unsigned)(ga & 0xFFFFFFFFu);                      // global_addr[31:0]
  g0[3] = (unsigned)((ga >> 32) & 0x1FFFFFFu) | (2u << 30);  // global_addr[56:32] | type=2
  i32x8 g1;
  g1[0] = (int)(1u << 16);                                   // data_size=1 -> 2 bytes/elem
  g1[1] = (int)((dim0 & 0xFFFFu) << 16);                     // tensor_dim0[15:0] at bits 63:48
  g1[2] = (int)(((dim0 >> 16) & 0xFFFFu) | ((dim1 & 0xFFFFu) << 16));
  g1[3] = (int)(((dim1 >> 16) & 0xFFFFu) | ((tile_w & 0xFFFFu) << 16)); // tile_dim0
  g1[4] = (int)(tile_h & 0xFFFFu);                           // tile_dim1 (tile_dim2 = 0)
  g1[5] = (int)stride0;                                      // tensor_dim0_stride[31:0]
  g1[6] = 0;
  g1[7] = 0;
  const i32x4 gz = {0, 0, 0, 0};
#if __clang_major__ >= 23
  const i32x8 gz8 = {0, 0, 0, 0, 0, 0, 0, 0};
  __builtin_amdgcn_tensor_load_to_lds(g0, g1, gz, gz, gz8, 0);
#else
  __builtin_amdgcn_tensor_load_to_lds(g0, g1, gz, gz, 0);
#endif
}

// ---------------------------------------------------------------- utilities
__global__ void k_zero(float* __restrict__ p, int n) {
  int i = blockIdx.x * blockDim.x + threadIdx.x;
  if (i < n) p[i] = 0.0f;
}

__global__ void k_cvt(_Float16* __restrict__ dst, const float* __restrict__ src, int n) {
  for (int i = blockIdx.x * blockDim.x + threadIdx.x; i < n; i += gridDim.x * blockDim.x)
    dst[i] = (_Float16)src[i];
}

// ---------------------------------------------------------------- CMS GEMM1: h = relu(buf16 @ W1^T + b1)
// Block tile 32(M) x 256(N); 8 waves as 2x4; register-pipelined K loop: fragments for
// step k+1 are requested before the WMMAs of step k so loads stay in flight across compute.
__global__ __launch_bounds__(256) void k_cms1(const int* gstep, int freq,
    const _Float16* __restrict__ buf16, const _Float16* __restrict__ W1,
    const float* __restrict__ b1, _Float16* __restrict__ h16)
{
  if ((gstep[0] % freq) != 0) return;
  const int tid = threadIdx.x, lane = tid & 31, w = tid >> 5;
  const int wm = w >> 2, wn = w & 3;
  const int mblk = blockIdx.x * 32;
  const int n0 = blockIdx.y * 256 + wn * 64;
  const int mrow = mblk + wm * 16;
  v8f acc[4] = {{}, {}, {}, {}};
  v16h a = load_a_frag_global(buf16, mrow, 0, D_INx, lane);
  v16h b0 = load_b_frag_global(W1, n0 +  0, 0, D_INx, lane);
  v16h b1f= load_b_frag_global(W1, n0 + 16, 0, D_INx, lane);
  v16h b2 = load_b_frag_global(W1, n0 + 32, 0, D_INx, lane);
  v16h b3 = load_b_frag_global(W1, n0 + 48, 0, D_INx, lane);
  for (int k0 = 0; k0 < D_INx - 32; k0 += 32) {
    const int kn = k0 + 32;
    v16h an  = load_a_frag_global(buf16, mrow, kn, D_INx, lane);
    v16h bn0 = load_b_frag_global(W1, n0 +  0, kn, D_INx, lane);
    v16h bn1 = load_b_frag_global(W1, n0 + 16, kn, D_INx, lane);
    v16h bn2 = load_b_frag_global(W1, n0 + 32, kn, D_INx, lane);
    v16h bn3 = load_b_frag_global(W1, n0 + 48, kn, D_INx, lane);
    acc[0] = wmma16x16x32(a, b0,  acc[0]);
    acc[1] = wmma16x16x32(a, b1f, acc[1]);
    acc[2] = wmma16x16x32(a, b2,  acc[2]);
    acc[3] = wmma16x16x32(a, b3,  acc[3]);
    a = an; b0 = bn0; b1f = bn1; b2 = bn2; b3 = bn3;
  }
  acc[0] = wmma16x16x32(a, b0,  acc[0]);
  acc[1] = wmma16x16x32(a, b1f, acc[1]);
  acc[2] = wmma16x16x32(a, b2,  acc[2]);
  acc[3] = wmma16x16x32(a, b3,  acc[3]);
  const int hi = lane >> 4;
  for (int s = 0; s < 4; ++s) {
    const int n = n0 + s * 16 + (lane & 15);
    const float bv = b1[n];
    for (int r = 0; r < 8; ++r) {
      const int m = mblk + wm * 16 + r + 8 * hi;
      float v = acc[s][r] + bv;
      h16[(size_t)m * HIDx + n] = (_Float16)(v > 0.0f ? v : 0.0f);
    }
  }
}

// ---------------------------------------------------------------- CMS GEMM2: buf = h @ W2^T + b2 + buf  (TDM-staged A)
__global__ __launch_bounds__(256) void k_cms2(const int* gstep, int freq,
    const _Float16* __restrict__ h16, const _Float16* __restrict__ W2,
    const float* __restrict__ b2, float* __restrict__ buf, _Float16* __restrict__ buf16)
{
  if ((gstep[0] % freq) != 0) return;
  __shared__ __align__(16) _Float16 Atile[2][32][32];
  const int tid = threadIdx.x, lane = tid & 31, w = tid >> 5;
  const int wm = w >> 2, wn = w & 3;
  const int mblk = blockIdx.x * 32;
  const int n0 = blockIdx.y * 128 + wn * 32;
  const int NK = HIDx / 32;
  if (w == 0)
    tdm_load_tile_f16((unsigned)(uintptr_t)&Atile[0][0][0],
                      h16 + (size_t)mblk * HIDx, HIDx, B_SZ, HIDx, 32, 32);
  v8f acc[2] = {{}, {}};
  for (int ki = 0; ki < NK; ++ki) {
    if (w == 0) __builtin_amdgcn_s_wait_tensorcnt(0);  // tile ki landed (issuing wave)
    __syncthreads();                                   // visible to all waves
    if (w == 0 && ki + 1 < NK)                         // DMA tile ki+1 while computing ki
      tdm_load_tile_f16((unsigned)(uintptr_t)&Atile[(ki + 1) & 1][0][0],
                        h16 + (size_t)mblk * HIDx + (ki + 1) * 32, HIDx, B_SZ, HIDx, 32, 32);
    const int k0 = ki * 32;
    v16h a = load_a_frag_lds(&Atile[ki & 1][wm * 16][0], lane, 32);
    acc[0] = wmma16x16x32(a, load_b_frag_global(W2, n0,      k0, HIDx, lane), acc[0]);
    acc[1] = wmma16x16x32(a, load_b_frag_global(W2, n0 + 16, k0, HIDx, lane), acc[1]);
    __syncthreads();                                   // reads done before buffer refill
  }
  const int hi = lane >> 4;
  for (int s = 0; s < 2; ++s) {
    const int n = n0 + s * 16 + (lane & 15);
    const float bv = b2[n];
    for (int r = 0; r < 8; ++r) {
      const size_t idx = (size_t)(mblk + wm * 16 + r + 8 * hi) * D_INx + n;
      const float v = acc[s][r] + bv + buf[idx];
      buf[idx] = v;
      buf16[idx] = (_Float16)v;   // keep f16 mirror coherent for downstream GEMMs
    }
  }
}

// ---------------------------------------------------------------- fused 3-way gate hidden
__device__ __forceinline__ void gates_step(v16h a, int k0, int n0, int lane,
    const _Float16* Wm, const _Float16* Wsn, const _Float16* Wg,
    v8f* am, v8f* asn, v8f* ag)
{
  for (int s = 0; s < 2; ++s) {
    am[s]  = wmma16x16x32(a, load_b_frag_global(Wm,  n0 + s * 16, k0, D_INx, lane), am[s]);
    asn[s] = wmma16x16x32(a, load_b_frag_global(Wsn, n0 + s * 16, k0, D_INx, lane), asn[s]);
    ag[s]  = wmma16x16x32(a, load_b_frag_global(Wg,  n0 + s * 16, k0, D_INx, lane), ag[s]);
  }
}

__global__ __launch_bounds__(256) void k_gates_h(
    const _Float16* __restrict__ buf16,
    const _Float16* __restrict__ Wm, const _Float16* __restrict__ Wsn, const _Float16* __restrict__ Wg,
    const float* __restrict__ bm, const float* __restrict__ bs, const float* __restrict__ bg,
    _Float16* __restrict__ sumH)
{
  const int tid = threadIdx.x, lane = tid & 31, w = tid >> 5;
  const int wm = w >> 2, wn = w & 3;
  const int mblk = blockIdx.x * 32;
  const int n0 = blockIdx.y * 128 + wn * 32;
  const int mrow = mblk + wm * 16;
  v8f am[2] = {{}, {}}, asn[2] = {{}, {}}, ag[2] = {{}, {}};
  v16h a = load_a_frag_global(buf16, mrow, 0, D_INx, lane);
  for (int k0 = 0; k0 < D_INx - 32; k0 += 32) {
    v16h an = load_a_frag_global(buf16, mrow, k0 + 32, D_INx, lane);  // prefetch next A
    gates_step(a, k0, n0, lane, Wm, Wsn, Wg, am, asn, ag);
    a = an;
  }
  gates_step(a, D_INx - 32, n0, lane, Wm, Wsn, Wg, am, asn, ag);
  const int hi = lane >> 4;
  for (int s = 0; s < 2; ++s) {
    const int n = n0 + s * 16 + (lane & 15);
    const float bmv = bm[n], bsv = bs[n], bgv = bg[n];
    for (int r = 0; r < 8; ++r) {
      const int m = mblk + wm * 16 + r + 8 * hi;
      float v = sigmoidf(am[s][r] + bmv) + sigmoidf(asn[s][r] + bsv) + sigmoidf(ag[s][r] + bgv);
      sumH[(size_t)m * HIDx + n] = (_Float16)v;
    }
  }
}

// ---------------------------------------------------------------- gates head: gates[b,0:3] = sigmoid(sumH @ gow^T + gob)
__global__ __launch_bounds__(256) void k_gates_out(
    const _Float16* __restrict__ sumH, const float* __restrict__ gow,
    const float* __restrict__ gob, float* __restrict__ gates, float* __restrict__ ssum)
{
  __shared__ float blk[3];
  const int tid = threadIdx.x, lane = tid & 31, w = tid >> 5;
  if (tid < 3) blk[tid] = 0.0f;
  __syncthreads();
  const int row = blockIdx.x * 8 + w;
  const int base = lane * 32;
  __align__(16) _Float16 hh[32];
  const _Float16* hp = &sumH[(size_t)row * HIDx + base];
  *(v8h*)&hh[0]  = *(const v8h*)(hp + 0);
  *(v8h*)&hh[8]  = *(const v8h*)(hp + 8);
  *(v8h*)&hh[16] = *(const v8h*)(hp + 16);
  *(v8h*)&hh[24] = *(const v8h*)(hp + 24);
  float a0 = 0.f, a1 = 0.f, a2 = 0.f;
  for (int i = 0; i < 32; ++i) {
    float hv = (float)hh[i];
    a0 += hv * gow[0 * HIDx + base + i];
    a1 += hv * gow[1 * HIDx + base + i];
    a2 += hv * gow[2 * HIDx + base + i];
  }
  for (int off = 16; off; off >>= 1) {
    a0 += __shfl_xor(a0, off, 32);
    a1 += __shfl_xor(a1, off, 32);
    a2 += __shfl_xor(a2, off, 32);
  }
  if (lane == 0) {
    float m = sigmoidf(a0 + gob[0]);
    float s = sigmoidf(a1 + gob[1]);
    float g = sigmoidf(a2 + gob[2]);
    gates[(size_t)row * 3 + 0] = m;
    gates[(size_t)row * 3 + 1] = s;
    gates[(size_t)row * 3 + 2] = g;
    atomicAdd(&blk[0], m); atomicAdd(&blk[1], s); atomicAdd(&blk[2], g);
  }
  __syncthreads();
  if (tid < 3) atomicAdd(&ssum[tid], blk[tid]);
}

// ---------------------------------------------------------------- fused slow/fast GEMM + gate + beta-SiLU + LayerNorm
__device__ __forceinline__ void actln_step(v16h a, int k0, int w, int lane,
    const _Float16* Wsl, const _Float16* Wfa, v8f* sa, v8f* fa)
{
  for (int s = 0; s < 4; ++s) {
    const int n = w * 64 + s * 16;
    sa[s] = wmma16x16x32(a, load_b_frag_global(Wsl, n, k0, D_INx, lane), sa[s]);
    fa[s] = wmma16x16x32(a, load_b_frag_global(Wfa, n, k0, D_INx, lane), fa[s]);
  }
}

__global__ __launch_bounds__(256) void k_actln(
    const _Float16* __restrict__ buf16, const _Float16* __restrict__ Wsl, const _Float16* __restrict__ Wfa,
    const float* __restrict__ gates, const float* __restrict__ gam, const float* __restrict__ bet,
    _Float16* __restrict__ fast16, float* __restrict__ sumsq, float* __restrict__ out)
{
  __shared__ float redS[8][16], redQ[8][16];
  __shared__ float grow[16], srow[16], muL[16], rsL[16];
  const int tid = threadIdx.x, lane = tid & 31, w = tid >> 5, hi = lane >> 4;
  const int mblk = blockIdx.x * 16;
  v8f sa[4] = {{}, {}, {}, {}}, fa[4] = {{}, {}, {}, {}};
  v16h a = load_a_frag_global(buf16, mblk, 0, D_INx, lane);
  for (int k0 = 0; k0 < D_INx - 32; k0 += 32) {
    v16h an = load_a_frag_global(buf16, mblk, k0 + 32, D_INx, lane);  // prefetch next A
    actln_step(a, k0, w, lane, Wsl, Wfa, sa, fa);
    a = an;
  }
  actln_step(a, D_INx - 32, w, lane, Wsl, Wfa, sa, fa);
  if (tid < 16) {
    grow[tid] = gates[(size_t)(mblk + tid) * 3 + 2];   // gate
    srow[tid] = gates[(size_t)(mblk + tid) * 3 + 1];   // sens
  }
  __syncthreads();
  float actv[4][8];
  float rowS[8], rowQ[8];
  for (int r = 0; r < 8; ++r) { rowS[r] = 0.f; rowQ[r] = 0.f; }
  for (int s = 0; s < 4; ++s) {
    const int n = w * 64 + s * 16 + (lane & 15);
    float fsq = 0.f;
    for (int r = 0; r < 8; ++r) {
      const int m = r + 8 * hi;
      const float fv = fa[s][r], sv = sa[s][r];
      const float comb = sv + fv * grow[m];
      const float beta = 0.5f + 2.0f * srow[m];
      const float act = comb * sigmoidf(beta * comb);
      actv[s][r] = act;
      rowS[r] += act; rowQ[r] += act * act;
      fsq += fv * fv;
      fast16[(size_t)(mblk + m) * D_OUTx + n] = (_Float16)fv;
    }
    fsq += __shfl_xor(fsq, 16, 32);
    if (hi == 0) atomicAdd(&sumsq[n], fsq);
  }
  for (int r = 0; r < 8; ++r)
    for (int off = 1; off < 16; off <<= 1) {
      rowS[r] += __shfl_xor(rowS[r], off, 32);
      rowQ[r] += __shfl_xor(rowQ[r], off, 32);
    }
  if ((lane & 15) == 0)
    for (int r = 0; r < 8; ++r) { redS[w][r + 8 * hi] = rowS[r]; redQ[w][r + 8 * hi] = rowQ[r]; }
  __syncthreads();
  if (tid < 16) {
    float S = 0.f, Q = 0.f;
    for (int ww = 0; ww < 8; ++ww) { S += redS[ww][tid]; Q += redQ[ww][tid]; }
    const float mu = S * (1.0f / D_OUTx);
    const float var = Q * (1.0f / D_OUTx) - mu * mu;
    muL[tid] = mu; rsL[tid] = rsqrtf(var + LN_EPS);
  }
  __syncthreads();
  for (int s = 0; s < 4; ++s) {
    const int n = w * 64 + s * 16 + (lane & 15);
    const float ga = gam[n], be = bet[n];
    for (int r = 0; r < 8; ++r) {
      const int m = r + 8 * hi;
      out[(size_t)(mblk + m) * D_OUTx + n] = (actv[s][r] - muL[m]) * rsL[m] * ga + be;
    }
  }
}

// ---------------------------------------------------------------- hebb = fast^T @ x_cms, split-K, LDS-transposed tiles
__global__ __launch_bounds__(128) void k_hebb(const _Float16* __restrict__ fast16,
    const _Float16* __restrict__ buf16, float* __restrict__ hebb)
{
  __shared__ __align__(16) _Float16 At[32][32];
  __shared__ __align__(16) _Float16 Bt[32][32];
  const int tid = threadIdx.x, lane = tid & 31, w = tid >> 5;
  const int wm = w >> 1, wn = w & 1;
  const int o0 = blockIdx.x * 32, d0 = blockIdx.y * 32;
  const int kbase = blockIdx.z * (B_SZ / 8);
  v8f acc = {};
  for (int ks = 0; ks < B_SZ / 8; ks += 32) {
    __syncthreads();
    { const int bl = tid >> 2, gi = tid & 3;
      const size_t b = (size_t)(kbase + ks + bl);
      v8h fv = *(const v8h*)&fast16[b * D_OUTx + o0 + gi * 8];
      v8h xv = *(const v8h*)&buf16[b * D_INx + d0 + gi * 8];
      for (int i = 0; i < 8; ++i) At[gi * 8 + i][bl] = fv[i];
      for (int i = 0; i < 8; ++i) Bt[gi * 8 + i][bl] = xv[i]; }
    __syncthreads();
    v16h a = load_a_frag_lds(&At[wm * 16][0], lane, 32);
    v16h b = load_b_frag_lds(&Bt[wn * 16][0], lane, 32);
    acc = wmma16x16x32(a, b, acc);
  }
  const int hi = lane >> 4;
  for (int r = 0; r < 8; ++r) {
    const int o = o0 + wm * 16 + r + 8 * hi;
    const int d = d0 + wn * 16 + (lane & 15);
    atomicAdd(&hebb[(size_t)o * D_INx + d], acc[r]);
  }
}

// ---------------------------------------------------------------- finalize: new W_fast + stats
__global__ __launch_bounds__(256) void k_final(const float* __restrict__ hebb,
    const float* __restrict__ sumsq, const float* __restrict__ ssum,
    const float* __restrict__ Wf, float* __restrict__ outW, float* __restrict__ outS)
{
  const int i = blockIdx.x * 256 + threadIdx.x;
  if (i >= D_OUTx * D_INx) return;
  const float invB = 1.0f / (float)B_SZ;
  const int o = i >> 9;
  const float rate = ssum[0] * invB * BASE_LR;
  const float wf = Wf[i];
  const float hb = hebb[i] * invB;
  const float forget = sumsq[o] * invB * wf;
  outW[i] = wf + tanhf(hb - forget) * rate;
  if (i < 3) outS[i] = ssum[i] * invB;
}

// ---------------------------------------------------------------- host
extern "C" void kernel_launch(void* const* d_in, const int* in_sizes, int n_in,
                              void* d_out, int out_size, void* d_ws, size_t ws_size,
                              hipStream_t stream)
{
  const float* x        = (const float*)d_in[0];
  const int*   gstep    = (const int*)  d_in[1];
  const float* cms_w1   = (const float*)d_in[2];
  const float* cms_b1   = (const float*)d_in[3];
  const float* cms_w2   = (const float*)d_in[4];
  const float* cms_b2   = (const float*)d_in[5];
  const float* g_metab_w= (const float*)d_in[6];
  const float* g_metab_b= (const float*)d_in[7];
  const float* g_sens_w = (const float*)d_in[8];
  const float* g_sens_b = (const float*)d_in[9];
  const float* g_gate_w = (const float*)d_in[10];
  const float* g_gate_b = (const float*)d_in[11];
  const float* g_out_w  = (const float*)d_in[12];
  const float* g_out_b  = (const float*)d_in[13];
  const float* W_slow   = (const float*)d_in[14];
  const float* W_fast   = (const float*)d_in[15];
  const float* ln_gamma = (const float*)d_in[16];
  const float* ln_beta  = (const float*)d_in[17];

  // workspace layout (bytes)
  const size_t OFF_BUF   = 0;                                          // f32  B x 512
  const size_t OFF_BUF16 = OFF_BUF   + (size_t)B_SZ * D_INx * 4;       // f16  B x 512 (mirror)
  const size_t OFF_H16   = OFF_BUF16 + (size_t)B_SZ * D_INx * 2;       // f16  B x 1024 (cms hidden / sumH)
  const size_t OFF_FAST  = OFF_H16   + (size_t)B_SZ * HIDx * 2;        // f16  B x 512
  const size_t OFF_GATES = OFF_FAST  + (size_t)B_SZ * D_OUTx * 2;      // f32  B x 3
  const size_t OFF_W16   = OFF_GATES + 1048576;                        // f16 weight pool (10.5 MB)
  const size_t OFF_HEBB  = OFF_W16   + 10485760;                       // f32 512x512
  const size_t OFF_SUMSQ = OFF_HEBB  + (size_t)D_OUTx * D_INx * 4;     // f32 512
  const size_t OFF_SSUM  = OFF_SUMSQ + 2048;                           // f32 8

  char* ws = (char*)d_ws;
  float*    buf    = (float*)   (ws + OFF_BUF);
  _Float16* buf16  = (_Float16*)(ws + OFF_BUF16);
  _Float16* h16    = (_Float16*)(ws + OFF_H16);
  _Float16* fast16 = (_Float16*)(ws + OFF_FAST);
  float*    gates  = (float*)   (ws + OFF_GATES);
  _Float16* w16    = (_Float16*)(ws + OFF_W16);
  float*    hebb   = (float*)   (ws + OFF_HEBB);
  float*    sumsq  = (float*)   (ws + OFF_SUMSQ);
  float*    ssum   = (float*)   (ws + OFF_SSUM);

  // f16 weight pool sub-offsets (in halves)
  const size_t W1OFF = 0;                          // 3 x 1024 x 512
  const size_t W2OFF = W1OFF + 3u * HIDx * D_INx;  // 3 x 512 x 1024
  const size_t GMOFF = W2OFF + 3u * D_INx * HIDx;  // 1024 x 512
  const size_t GSOFF = GMOFF + (size_t)HIDx * D_INx;
  const size_t GGOFF = GSOFF + (size_t)HIDx * D_INx;
  const size_t WSOFF = GGOFF + (size_t)HIDx * D_INx;   // 512 x 512
  const size_t WFOFF = WSOFF + (size_t)D_OUTx * D_INx; // 512 x 512

  // 0) buf = x (f32), buf16 = f16 mirror, zero accumulators
  hipMemcpyAsync(buf, x, (size_t)B_SZ * D_INx * 4, hipMemcpyDeviceToDevice, stream);
  k_cvt<<<2048, 256, 0, stream>>>(buf16, x, B_SZ * D_INx);
  const int nzero = (D_OUTx * D_INx) + 512 + 8;
  k_zero<<<(nzero + 255) / 256, 256, 0, stream>>>(hebb, nzero);

  // 1) convert all GEMM weights to f16 (~10.5 MB, stays resident in the 192MB L2)
  k_cvt<<<1024, 256, 0, stream>>>(w16 + W1OFF, cms_w1,   3 * HIDx * D_INx);
  k_cvt<<<1024, 256, 0, stream>>>(w16 + W2OFF, cms_w2,   3 * D_INx * HIDx);
  k_cvt<<<1024, 256, 0, stream>>>(w16 + GMOFF, g_metab_w, HIDx * D_INx);
  k_cvt<<<1024, 256, 0, stream>>>(w16 + GSOFF, g_sens_w,  HIDx * D_INx);
  k_cvt<<<1024, 256, 0, stream>>>(w16 + GGOFF, g_gate_w,  HIDx * D_INx);
  k_cvt<<<1024, 256, 0, stream>>>(w16 + WSOFF, W_slow,    D_OUTx * D_INx);
  k_cvt<<<1024, 256, 0, stream>>>(w16 + WFOFF, W_fast,    D_OUTx * D_INx);

  // 2) CMS residual MLP levels (predicated on global_step % freq on-device)
  const int freqs[3] = {1, 4, 16};
  for (int l = 0; l < 3; ++l) {
    k_cms1<<<dim3(B_SZ / 32, HIDx / 256), 256, 0, stream>>>(
        gstep, freqs[l], buf16, w16 + W1OFF + (size_t)l * HIDx * D_INx, cms_b1 + l * HIDx, h16);
    k_cms2<<<dim3(B_SZ / 32, D_INx / 128), 256, 0, stream>>>(
        gstep, freqs[l], h16, w16 + W2OFF + (size_t)l * D_INx * HIDx, cms_b2 + l * D_INx, buf, buf16);
  }

  // 3) fused 3-way gate hidden -> sumH (reuses h16)
  k_gates_h<<<dim3(B_SZ / 32, HIDx / 128), 256, 0, stream>>>(
      buf16, w16 + GMOFF, w16 + GSOFF, w16 + GGOFF, g_metab_b, g_sens_b, g_gate_b, h16);

  // 4) gates head + stat sums
  k_gates_out<<<B_SZ / 8, 256, 0, stream>>>(h16, g_out_w, g_out_b, gates, ssum);

  // 5) slow/fast GEMM + gating + beta-SiLU + LayerNorm -> ln_out; emits fast16 and sum(fast^2)
  float* out_ln = (float*)d_out;
  float* out_wf = out_ln + (size_t)B_SZ * D_OUTx;
  float* out_st = out_wf + (size_t)D_OUTx * D_INx;
  k_actln<<<B_SZ / 16, 256, 0, stream>>>(
      buf16, w16 + WSOFF, w16 + WFOFF, gates, ln_gamma, ln_beta, fast16, sumsq, out_ln);

  // 6) hebb = fast^T @ x_cms (split-K=8, atomic accumulate)
  k_hebb<<<dim3(D_OUTx / 32, D_INx / 32, 8), 128, 0, stream>>>(fast16, buf16, hebb);

  // 7) finalize: new W_fast + stats
  k_final<<<(D_OUTx * D_INx + 255) / 256, 256, 0, stream>>>(hebb, sumsq, ssum, W_fast, out_wf, out_st);
}
